// MXCrossAttention_45775761441070
// MI455X (gfx1250) — matrix operand order repaired and checked by reading, not compiled, not run on hardware
//
#include <hip/hip_runtime.h>
#include <hip/hip_bf16.h>
#include <stdint.h>

// ---------------------------------------------------------------------------
// Types for CDNA5 WMMA (wave32): bf16 A/B fragments, f32 accumulators.
// ---------------------------------------------------------------------------
typedef __bf16 bf16;
typedef __bf16 v16bf __attribute__((ext_vector_type(16)));
typedef __bf16 v8bf  __attribute__((ext_vector_type(8)));
typedef float  v8f   __attribute__((ext_vector_type(8)));

#define DEVINL __device__ __forceinline__

DEVINL uint16_t f2bfu(float f) {
  uint32_t u = __builtin_bit_cast(uint32_t, f);
  return (uint16_t)((u + 0x7FFFu + ((u >> 16) & 1u)) >> 16);  // round-nearest-even
}
DEVINL bf16 f2bf(float f) {
  uint16_t s = f2bfu(f);
  return __builtin_bit_cast(bf16, s);
}
DEVINL v16bf cat16(v8bf lo, v8bf hi) {
  return __builtin_shufflevector(lo, hi, 0,1,2,3,4,5,6,7,8,9,10,11,12,13,14,15);
}
DEVINL v8f wmma_bf16(v16bf a, v16bf b, v8f c) {
  return __builtin_amdgcn_wmma_f32_16x16x32_bf16(false, a, false, b, (short)0, c,
                                                 false, false);
}

// Problem constants (from reference): B=4, N=4096, T=77, C=1024, H=16, D=64
#define BB 4
#define NN 4096
#define TT 77
#define TP 96          // T padded to 3 K-steps of 32
#define CC 1024
#define HH 16
#define DD 64

// ---------------------------------------------------------------------------
// fp32 -> bf16 conversion (4 elems / thread, 8B packed stores)
// ---------------------------------------------------------------------------
__global__ __launch_bounds__(256) void cvt_f32_bf16_k(const float* __restrict__ in,
                                                      uint64_t* __restrict__ out,
                                                      long long n4) {
  long long i = (long long)blockIdx.x * blockDim.x + threadIdx.x;
  long long stride = (long long)gridDim.x * blockDim.x;
  for (; i < n4; i += stride) {
    float4 v = ((const float4*)in)[i];
    uint64_t p = (uint64_t)f2bfu(v.x) | ((uint64_t)f2bfu(v.y) << 16) |
                 ((uint64_t)f2bfu(v.z) << 32) | ((uint64_t)f2bfu(v.w) << 48);
    out[i] = p;
  }
}

__global__ __launch_bounds__(256) void zero_k(uint32_t* __restrict__ p, long long n32) {
  long long i = (long long)blockIdx.x * blockDim.x + threadIdx.x;
  long long stride = (long long)gridDim.x * blockDim.x;
  for (; i < n32; i += stride) p[i] = 0u;
}

// ---------------------------------------------------------------------------
// WMMA GEMM: Out = A(MxK) * W(NxK)^T + bias,  K == 1024, N == 1024.
// Block tile 128x128, BK=32, 256 threads = 8 waves arranged 2(m) x 4(n);
// each wave owns a 64x32 tile = 4x2 WMMA accumulators. Ping-pong LDS
// double-buffering: one barrier per K-step, global loads for tile i+1
// overlap the WMMAs of tile i; global_prefetch issued 2 tiles ahead.
// MODE 0: bf16 out, (acc+bias)*scale           (Q projection, scale=1/8)
// MODE 1: bf16 out remapped to [B,H,TP,D]      (K/V projections, M=B*T)
// MODE 2: f32 out, acc+bias                    (output projection)
// ---------------------------------------------------------------------------
template <int MODE>
__global__ __launch_bounds__(256) void wmma_gemm_xwT(
    const bf16* __restrict__ A, const bf16* __restrict__ W,
    const float* __restrict__ bias, void* __restrict__ Out, int M, float scale) {
  constexpr int K = CC;
  constexpr int KSTEPS = K / 32;
  constexpr int SA = 40;  // LDS row stride (elems): 80B, 16B-aligned
  __shared__ bf16 lA[2][128 * SA];
  __shared__ bf16 lB[2][128 * SA];

  const int tid  = threadIdx.x;
  const int lane = tid & 31;
  const int w    = tid >> 5;
  const int wm   = w >> 2;      // 0..1
  const int wn   = w & 3;       // 0..3
  const int half = lane >> 4;   // 0..1
  const int l16  = lane & 15;
  const int mBase = blockIdx.x * 128;
  const int nBase = blockIdx.y * 128;

  const int fr = tid >> 1;            // fill row 0..127
  const int fc = (tid & 1) * 16;      // fill col chunk 0 / 16

  const v8bf zro = {};
  const bool aValid = (mBase + fr) < M;
  const bf16* aPtr = A + (size_t)(mBase + fr) * K + fc;  // only deref if aValid
  const bf16* wPtr = W + (size_t)(nBase + fr) * K + fc;

  // ---- prologue: stage tile 0 into buffer 0 ----
  v8bf ra0 = zro, ra1 = zro, rb0, rb1;
  if (aValid) { ra0 = *(const v8bf*)aPtr; ra1 = *(const v8bf*)(aPtr + 8); }
  rb0 = *(const v8bf*)wPtr; rb1 = *(const v8bf*)(wPtr + 8);
  *(v8bf*)&lA[0][fr * SA + fc]     = ra0;
  *(v8bf*)&lA[0][fr * SA + fc + 8] = ra1;
  *(v8bf*)&lB[0][fr * SA + fc]     = rb0;
  *(v8bf*)&lB[0][fr * SA + fc + 8] = rb1;
  __syncthreads();

  v8f acc[4][2] = {};

  for (int it = 0; it < KSTEPS; ++it) {
    const int buf = it & 1;
    const bool hasNext = (it + 1) < KSTEPS;

    // ---- kick off global loads for tile it+1 (land after the WMMAs) ----
    if (hasNext) {
      const int ko = (it + 1) * 32;
      ra0 = zro; ra1 = zro;
      if (aValid) {
        ra0 = *(const v8bf*)(aPtr + ko);
        ra1 = *(const v8bf*)(aPtr + ko + 8);
      }
      rb0 = *(const v8bf*)(wPtr + ko);
      rb1 = *(const v8bf*)(wPtr + ko + 8);
      if (it + 2 < KSTEPS) {  // gfx1250 global_prefetch two tiles ahead
        __builtin_prefetch(aPtr + (it + 2) * 32, 0, 1);
        __builtin_prefetch(wPtr + (it + 2) * 32, 0, 1);
      }
    }

    // ---- compute from current buffer ----
    const bf16* baseA = lA[buf];
    const bf16* baseB = lB[buf];
    v16bf af[4], bfr[2];
#pragma unroll
    for (int i = 0; i < 4; ++i) {
      // A frag: lane holds row (l16), K chunks half*8 and 16+half*8
      const bf16* p = &baseA[(wm * 64 + i * 16 + l16) * SA + half * 8];
      af[i] = cat16(*(const v8bf*)p, *(const v8bf*)(p + 16));
    }
#pragma unroll
    for (int j = 0; j < 2; ++j) {
      // B frag: lane holds column (l16) of W^T = row of W, K chunks half*8 and 16+half*8
      const bf16* p = &baseB[(wn * 32 + j * 16 + l16) * SA + half * 8];
      bfr[j] = cat16(*(const v8bf*)p, *(const v8bf*)(p + 16));
    }
#pragma unroll
    for (int i = 0; i < 4; ++i)
#pragma unroll
      for (int j = 0; j < 2; ++j) acc[i][j] = wmma_bf16(af[i], bfr[j], acc[i][j]);

    // ---- store tile it+1 into the other buffer (read-safe: barrier at end
    //      of previous iteration guaranteed all reads of buf^1 finished) ----
    if (hasNext) {
      bf16* dA = lA[buf ^ 1];
      bf16* dB = lB[buf ^ 1];
      *(v8bf*)&dA[fr * SA + fc]     = ra0;
      *(v8bf*)&dA[fr * SA + fc + 8] = ra1;
      *(v8bf*)&dB[fr * SA + fc]     = rb0;
      *(v8bf*)&dB[fr * SA + fc + 8] = rb1;
    }
    __syncthreads();
  }

  // Epilogue. D layout: lane l16 = column, rows = half*8 + m in vgpr m.
#pragma unroll
  for (int j = 0; j < 2; ++j) {
    int col = nBase + wn * 32 + j * 16 + l16;
    float bv = bias[col];
#pragma unroll
    for (int i = 0; i < 4; ++i) {
#pragma unroll
      for (int m = 0; m < 8; ++m) {
        int row = mBase + wm * 64 + i * 16 + half * 8 + m;
        if (row >= M) continue;
        float v = acc[i][j][m] + bv;
        if (MODE == 0) {
          ((bf16*)Out)[(size_t)row * CC + col] = f2bf(v * scale);
        } else if (MODE == 1) {
          int b = row / TT, t = row % TT;
          int h = col >> 6, d = col & 63;
          ((bf16*)Out)[(((size_t)(b * HH + h)) * TP + t) * DD + d] = f2bf(v);
        } else {
          ((float*)Out)[(size_t)row * CC + col] = v;
        }
      }
    }
  }
}

// ---------------------------------------------------------------------------
// Fused attention: per block = (b, h, 128 Q rows). K (TPx64) and V^T (64xTP)
// staged in LDS; scores + softmax + P*V all in one kernel, all matmuls WMMA.
// ---------------------------------------------------------------------------
__global__ __launch_bounds__(256) void attention_k(
    const bf16* __restrict__ Qb,   // [B,N,C] bf16, scale folded in
    const bf16* __restrict__ Kb,   // [B,H,TP,D] bf16 (rows >=77 zero)
    const bf16* __restrict__ Vb,   // [B,H,TP,D] bf16 (rows >=77 zero)
    const float* __restrict__ amask,  // [B,N,T] f32 additive mask
    bf16* __restrict__ Ctx) {      // [B,N,C] bf16
  constexpr int SK = 72;   // K LDS row stride (144B)
  constexpr int SV = 104;  // V^T LDS row stride (208B)
  constexpr int SP = 96;   // P LDS row stride (192B)
  __shared__ bf16 Ks[TP * SK];
  __shared__ bf16 Vt[DD * SV];
  __shared__ bf16 Ps[8 * 16 * SP];

  const int tid  = threadIdx.x;
  const int lane = tid & 31;
  const int w    = tid >> 5;
  const int half = lane >> 4;
  const int l16  = lane & 15;
  const int nt = blockIdx.x, h = blockIdx.y, b = blockIdx.z;
  const int nBase = nt * 128;

  const bf16* kbase = Kb + ((size_t)(b * HH + h)) * TP * DD;
  const bf16* vbase = Vb + ((size_t)(b * HH + h)) * TP * DD;

  // Stage K rows (row-major) and V transposed (so P*V B-frags are contiguous)
  for (int idx = tid; idx < TP * 4; idx += 256) {
    int r = idx >> 2, c = (idx & 3) * 16;
    *(v8bf*)&Ks[r * SK + c]     = *(const v8bf*)&kbase[r * DD + c];
    *(v8bf*)&Ks[r * SK + c + 8] = *(const v8bf*)&kbase[r * DD + c + 8];
  }
  for (int idx = tid; idx < TP * DD; idx += 256) {
    int t = idx >> 6, d = idx & 63;
    Vt[d * SV + t] = vbase[t * DD + d];
  }
  __syncthreads();

  // ---- scores: S[16 x TP] = Q_tile(16x64) * K^T, K-steps of 32 ----
  v8f sAcc[6] = {};
  const int qrow = nBase + w * 16 + l16;
  const bf16* qp = Qb + ((size_t)b * NN + qrow) * CC + h * DD;
#pragma unroll
  for (int kk = 0; kk < DD; kk += 32) {
    const bf16* ap = qp + kk + half * 8;
    v16bf a = cat16(*(const v8bf*)ap, *(const v8bf*)(ap + 16));
#pragma unroll
    for (int j = 0; j < 6; ++j) {
      const bf16* bp = &Ks[(j * 16 + l16) * SK + kk + half * 8];
      v16bf bb = cat16(*(const v8bf*)bp, *(const v8bf*)(bp + 16));
      sAcc[j] = wmma_bf16(a, bb, sAcc[j]);
    }
  }

  // ---- additive mask + pad masking; softmax over T in registers ----
  const int row0 = nBase + w * 16 + half * 8;  // first of my 8 rows
  float mx[8];
#pragma unroll
  for (int m = 0; m < 8; ++m) mx[m] = -1e30f;
#pragma unroll
  for (int j = 0; j < 6; ++j) {
    int t = j * 16 + l16;
    if (t < TT) {
      const float* mp = amask + ((size_t)b * NN + row0) * TT + t;
#pragma unroll
      for (int m = 0; m < 8; ++m) sAcc[j][m] += mp[(size_t)m * TT];
    } else {
#pragma unroll
      for (int m = 0; m < 8; ++m) sAcc[j][m] = -1e30f;
    }
#pragma unroll
    for (int m = 0; m < 8; ++m) mx[m] = fmaxf(mx[m], sAcc[j][m]);
  }
#pragma unroll
  for (int s = 1; s < 16; s <<= 1)
#pragma unroll
    for (int m = 0; m < 8; ++m) mx[m] = fmaxf(mx[m], __shfl_xor(mx[m], s, 32));

  float sum[8] = {0.f, 0.f, 0.f, 0.f, 0.f, 0.f, 0.f, 0.f};
#pragma unroll
  for (int j = 0; j < 6; ++j)
#pragma unroll
    for (int m = 0; m < 8; ++m) {
      float p = __expf(sAcc[j][m] - mx[m]);
      sAcc[j][m] = p;
      sum[m] += p;
    }
#pragma unroll
  for (int s = 1; s < 16; s <<= 1)
#pragma unroll
    for (int m = 0; m < 8; ++m) sum[m] += __shfl_xor(sum[m], s, 32);
  float inv[8];
#pragma unroll
  for (int m = 0; m < 8; ++m) inv[m] = 1.0f / sum[m];

  // ---- round-trip P through LDS to reshape D-layout -> A-fragment layout ----
  bf16* prow = &Ps[(w * 16) * SP];
#pragma unroll
  for (int j = 0; j < 6; ++j) {
    int t = j * 16 + l16;
#pragma unroll
    for (int m = 0; m < 8; ++m) prow[(half * 8 + m) * SP + t] = f2bf(sAcc[j][m]);
  }
  __syncthreads();

  // ---- out: O[16 x 64] = P(16 x TP) * V(TP x 64), K-steps of 32 ----
  v8f oAcc[4] = {};
#pragma unroll
  for (int kk = 0; kk < TP; kk += 32) {
    const bf16* ap = &Ps[(w * 16 + l16) * SP + kk + half * 8];
    v16bf a = cat16(*(const v8bf*)ap, *(const v8bf*)(ap + 16));
#pragma unroll
    for (int dt = 0; dt < 4; ++dt) {
      const bf16* bp = &Vt[(dt * 16 + l16) * SV + kk + half * 8];
      v16bf bb = cat16(*(const v8bf*)bp, *(const v8bf*)(bp + 16));
      oAcc[dt] = wmma_bf16(a, bb, oAcc[dt]);
    }
  }

  // ---- normalize by softmax sum and store context [B,N,C] ----
#pragma unroll
  for (int dt = 0; dt < 4; ++dt) {
    int col = h * DD + dt * 16 + l16;
#pragma unroll
    for (int m = 0; m < 8; ++m) {
      int row = row0 + m;
      Ctx[((size_t)b * NN + row) * CC + col] = f2bf(oAcc[dt][m] * inv[m]);
    }
  }
}

// ---------------------------------------------------------------------------
// Host-side orchestration
// ---------------------------------------------------------------------------
extern "C" void kernel_launch(void* const* d_in, const int* in_sizes, int n_in,
                              void* d_out, int out_size, void* d_ws, size_t ws_size,
                              hipStream_t stream) {
  (void)in_sizes; (void)n_in; (void)out_size; (void)ws_size;
  const float* hs  = (const float*)d_in[0];
  const float* ehs = (const float*)d_in[1];
  const float* msk = (const float*)d_in[2];
  const float* Wq  = (const float*)d_in[3];
  const float* bq  = (const float*)d_in[4];
  const float* Wk  = (const float*)d_in[5];
  const float* bk  = (const float*)d_in[6];
  const float* Wv  = (const float*)d_in[7];
  const float* bv  = (const float*)d_in[8];
  const float* Wo  = (const float*)d_in[9];
  const float* bo  = (const float*)d_in[10];

  char* ws = (char*)d_ws;
  size_t off = 0;
  auto take = [&](size_t bytes) {
    void* p = ws + off;
    off = (off + bytes + 255) & ~(size_t)255;
    return p;
  };
  const size_t MQ = (size_t)BB * NN;       // 16384
  const size_t MK = (size_t)BB * TT;       // 308
  bf16* hsB  = (bf16*)take(MQ * CC * 2);
  bf16* ehsB = (bf16*)take(MK * CC * 2);
  bf16* WqB  = (bf16*)take((size_t)CC * CC * 2);
  bf16* WkB  = (bf16*)take((size_t)CC * CC * 2);
  bf16* WvB  = (bf16*)take((size_t)CC * CC * 2);
  bf16* WoB  = (bf16*)take((size_t)CC * CC * 2);
  bf16* Qb   = (bf16*)take(MQ * CC * 2);
  bf16* Kb   = (bf16*)take((size_t)BB * HH * TP * DD * 2);
  bf16* Vb   = (bf16*)take((size_t)BB * HH * TP * DD * 2);
  bf16* Ctx  = (bf16*)take(MQ * CC * 2);

  // fp32 -> bf16 conversions
  auto cvt = [&](const float* in, bf16* out, long long n) {
    long long n4 = n / 4;
    int blocks = (int)((n4 + 255) / 256);
    if (blocks > 4096) blocks = 4096;
    cvt_f32_bf16_k<<<blocks, 256, 0, stream>>>(in, (uint64_t*)out, n4);
  };
  cvt(hs,  hsB,  (long long)MQ * CC);
  cvt(ehs, ehsB, (long long)MK * CC);
  cvt(Wq, WqB, (long long)CC * CC);
  cvt(Wk, WkB, (long long)CC * CC);
  cvt(Wv, WvB, (long long)CC * CC);
  cvt(Wo, WoB, (long long)CC * CC);

  // zero K/V padded buffers (Kb and Vb are adjacent in ws)
  {
    long long n32 = ((long long)BB * HH * TP * DD * 2 * 2) / 4;
    zero_k<<<1536, 256, 0, stream>>>((uint32_t*)Kb, n32);
  }

  const float qscale = 0.125f;  // 1/sqrt(D=64), folded into Q
  // Q = hs @ Wq^T + bq (scaled, bf16)
  wmma_gemm_xwT<0><<<dim3(128, 8), 256, 0, stream>>>(hsB, WqB, bq, Qb, (int)MQ, qscale);
  // K,V = ehs @ W^T + b, remapped to [B,H,TP,D]
  wmma_gemm_xwT<1><<<dim3(3, 8), 256, 0, stream>>>(ehsB, WkB, bk, Kb, (int)MK, 1.0f);
  wmma_gemm_xwT<1><<<dim3(3, 8), 256, 0, stream>>>(ehsB, WvB, bv, Vb, (int)MK, 1.0f);

  // fused softmax(QK^T + mask) V
  attention_k<<<dim3(NN / 128, HH, BB), 256, 0, stream>>>(Qb, Kb, Vb, msk, Ctx);

  // out = Ctx @ Wo^T + bo (f32)
  wmma_gemm_xwT<2><<<dim3(128, 8), 256, 0, stream>>>(Ctx, WoB, bo, (float*)d_out,
                                                     (int)MQ, 1.0f);
}
